// MultiHeadAttentionLayer_6597069767104
// MI455X (gfx1250) — compile-verified, matrix operand order checked by reading, not run on hardware
//
#include <hip/hip_runtime.h>
#include <hip/hip_bf16.h>

#define B_   2
#define L_   2048
#define HID_ 1024
#define H_   16
#define HD_  64

typedef __bf16 bf16_t;
typedef __attribute__((ext_vector_type(8)))  __bf16 v8bf;
typedef __attribute__((ext_vector_type(16))) __bf16 v16bf;
typedef __attribute__((ext_vector_type(8)))  float  v8f;

// Load a 16x32 bf16 fragment (A or B role) from a row-major source.
// Lane semantics (ISA 7.12.2, 16-bit A-matrix 16x32):
//   row  = lane%16 selects the M (or N) line
//   half = lane/16 : regs 0-3 hold K = half*8.., regs 4-7 hold K = 16+half*8..
static __device__ inline v16bf load_frag(const bf16_t* p, int half) {
    v8bf c0 = *(const v8bf*)(p + half * 8);
    v8bf c1 = *(const v8bf*)(p + 16 + half * 8);
    return __builtin_shufflevector(c0, c1, 0,1,2,3,4,5,6,7,8,9,10,11,12,13,14,15);
}

static __device__ inline v8f wmma_bf16(v16bf a, v16bf b, v8f c) {
    // v_wmma_f32_16x16x32_bf16  D = A*B + C  (f32 accumulate)
    return __builtin_amdgcn_wmma_f32_16x16x32_bf16(false, a, false, b, (short)0, c,
                                                   false, false);
}

// Async copy 16B from global to LDS (workgroup-relative byte offset), ASYNCcnt-tracked.
static __device__ inline void async_copy_b128(uint32_t lds_off, const void* gptr) {
    asm volatile("global_load_async_to_lds_b128 %0, %1, off"
                 :: "v"(lds_off), "v"((unsigned long long)(uintptr_t)gptr)
                 : "memory");
}

// ---------------------------------------------------------------- convert
__global__ __launch_bounds__(256) void f32_to_bf16_kernel(const float* __restrict__ in,
                                                          bf16_t* __restrict__ out,
                                                          size_t n) {
    size_t i = (size_t)blockIdx.x * blockDim.x + threadIdx.x;
    if (i < n) out[i] = (bf16_t)in[i];
}

// ---------------------------------------------------------------- tiled GEMM
// C[m,n] = sum_k X[m,k] * W[n,k] + bias[n]
// 8-wave workgroup computes a 128(M) x 128(N) macro-tile; K staged through LDS in
// double-buffered 64-deep chunks via GLOBAL_LOAD_ASYNC_TO_LDS_B128 (ASYNCcnt).
// Wave (wm = w&3, wn = w>>2) owns a 32(M) x 64(N) sub-tile -> 2x4 WMMA accumulators;
// per k-step: 6 LDS fragments feed 8 WMMAs (A and B both reused).
// mode 0: bf16 out, head-major  [B,H,L,HD]
// mode 1: bf16 out, d-major     [B,H,HD,L]   (V transposed for the PV pass)
// mode 2: f32 out, row-major    [M,HID]
#define MT 128
#define NT 128
#define KC 64
#define A_BYTES (MT * KC * 2)                 // 16 KB
#define B_BYTES (NT * KC * 2)                 // 16 KB
#define GEMM_SMEM (2 * (A_BYTES + B_BYTES))   // 64 KB double-buffered

__global__ __launch_bounds__(256) void tiled_gemm_kernel(const bf16_t* __restrict__ X,
                                                         const bf16_t* __restrict__ W,
                                                         const float*  __restrict__ bias,
                                                         void* __restrict__ Yv,
                                                         int K, int mode) {
    extern __shared__ char smem[];
    const int n0   = blockIdx.x * NT;
    const int m0   = blockIdx.y * MT;
    const int tid  = threadIdx.x;
    const int wave = tid >> 5;
    const int lane = tid & 31;
    const int row  = lane & 15;
    const int half = lane >> 4;
    const int wm   = wave & 3;    // M sub-tile (32 rows)
    const int wn   = wave >> 2;   // N half (64 cols)

    // Issue one KC-deep stage: 4 A-chunks + 4 B-chunks of 16B per thread, coalesced.
    auto issue_stage = [&](int kc, int bf) {
        const uint32_t aoff = (uint32_t)(bf * A_BYTES);
        const uint32_t boff = (uint32_t)(2 * A_BYTES + bf * B_BYTES);
#pragma unroll
        for (int u = 0; u < 4; ++u) {            // 1024 chunks of A: 128 rows x 8
            int c = u * 256 + tid;
            int r = c >> 3, q = c & 7;
            const bf16_t* src = X + (size_t)(m0 + r) * K + kc + q * 8;
            async_copy_b128(aoff + (uint32_t)(r * (KC * 2) + q * 16), src);
        }
#pragma unroll
        for (int u = 0; u < 4; ++u) {            // 1024 chunks of B: 128 rows x 8
            int c = u * 256 + tid;
            int r = c >> 3, q = c & 7;
            const bf16_t* src = W + (size_t)(n0 + r) * K + kc + q * 8;
            async_copy_b128(boff + (uint32_t)(r * (KC * 2) + q * 16), src);
        }
    };

    v8f acc[2][4] = {{{}, {}, {}, {}}, {{}, {}, {}, {}}};
    const int nStages = K / KC;
    issue_stage(0, 0);
    for (int s = 0; s < nStages; ++s) {
        if (s + 1 < nStages) {
            issue_stage((s + 1) * KC, (s + 1) & 1);
            // 8 outstanding per stage per wave; async loads complete in order,
            // so <=8 outstanding means stage s has fully landed.
            asm volatile("s_wait_asynccnt 8" ::: "memory");
        } else {
            asm volatile("s_wait_asynccnt 0" ::: "memory");
        }
        __syncthreads();   // stage s visible from every wave
        const bf16_t* sA = (const bf16_t*)(smem + (s & 1) * A_BYTES);
        const bf16_t* sB = (const bf16_t*)(smem + 2 * A_BYTES + (s & 1) * B_BYTES);
#pragma unroll
        for (int ks = 0; ks < KC; ks += 32) {
            // preload ALL fragments for this k-step so ds loads overlap the WMMAs
            v16bf a0 = load_frag(sA + (wm * 32 + row) * KC + ks, half);
            v16bf a1 = load_frag(sA + (wm * 32 + 16 + row) * KC + ks, half);
            v16bf b0 = load_frag(sB + (wn * 64 + 0 * 16 + row) * KC + ks, half);
            v16bf b1 = load_frag(sB + (wn * 64 + 1 * 16 + row) * KC + ks, half);
            v16bf b2 = load_frag(sB + (wn * 64 + 2 * 16 + row) * KC + ks, half);
            v16bf b3 = load_frag(sB + (wn * 64 + 3 * 16 + row) * KC + ks, half);
            acc[0][0] = wmma_bf16(a0, b0, acc[0][0]);
            acc[0][1] = wmma_bf16(a0, b1, acc[0][1]);
            acc[0][2] = wmma_bf16(a0, b2, acc[0][2]);
            acc[0][3] = wmma_bf16(a0, b3, acc[0][3]);
            acc[1][0] = wmma_bf16(a1, b0, acc[1][0]);
            acc[1][1] = wmma_bf16(a1, b1, acc[1][1]);
            acc[1][2] = wmma_bf16(a1, b2, acc[1][2]);
            acc[1][3] = wmma_bf16(a1, b3, acc[1][3]);
        }
        __syncthreads();   // all reads done before this buffer is overwritten
    }

    // Epilogue: C layout = N in lanes (lane%16), M = (lane/16)*8 + reg.
#pragma unroll
    for (int mm = 0; mm < 2; ++mm) {
#pragma unroll
        for (int t = 0; t < 4; ++t) {
            int n = n0 + wn * 64 + t * 16 + row;
            float bn = bias[n];
            int h = n >> 6, d = n & 63;
#pragma unroll
            for (int i = 0; i < 8; ++i) {
                int m = m0 + wm * 32 + mm * 16 + half * 8 + i;
                float v = acc[mm][t][i] + bn;
                if (mode == 2) {
                    ((float*)Yv)[(size_t)m * HID_ + n] = v;
                } else {
                    int b_ = m >> 11, l = m & (L_ - 1);
                    size_t off = (mode == 0)
                        ? ((((size_t)b_ * H_ + h) * L_ + l) * HD_ + d)
                        : ((((size_t)b_ * H_ + h) * HD_ + d) * L_ + l);
                    ((bf16_t*)Yv)[off] = (bf16_t)v;
                }
            }
        }
    }
}

// ---------------------------------------------------------------- rel-k scores
// R[b,h,l,j] = dot(Q[b,h,l,:], rel_k_table[j,:])   j in [0,5), stored padded to 8
__global__ __launch_bounds__(256) void relk_kernel(const bf16_t* __restrict__ Qh,
                                                   const float* __restrict__ relk,
                                                   float* __restrict__ R) {
    size_t t = (size_t)blockIdx.x * blockDim.x + threadIdx.x;
    const size_t total = (size_t)B_ * H_ * L_ * 5;
    if (t >= total) return;
    int j = (int)(t % 5);
    size_t rowid = t / 5;
    const bf16_t* q  = Qh + rowid * HD_;
    const float*  rk = relk + (size_t)j * HD_;
    float s = 0.f;
#pragma unroll 8
    for (int d = 0; d < HD_; ++d) s += (float)q[d] * rk[d];
    R[rowid * 8 + j] = s;
}

// ---------------------------------------------------------------- fused attention
// One workgroup = (b, h, 16 query rows). 8 waves split the 2048 keys.
// LDS: full 16x2048 bf16 score/prob row-block (CDNA5 320KB WGP LDS makes this cheap).
#define SMEM_BYTES (16 * L_ * 2 /*sP*/ + 4 * (8*16*64 /*sOp*/ + 16*8 /*sR*/ + 5*64 /*sRV*/ \
                    + 16 /*max*/ + 16 /*sum*/ + 256 /*sRed*/ + 16*16*5 /*sCp*/ + 16*5 /*sC*/))

__global__ __launch_bounds__(256) void attn_kernel(const bf16_t* __restrict__ Qh,   // [B,H,L,HD]
                                                   const bf16_t* __restrict__ Kh,   // [B,H,L,HD]
                                                   const bf16_t* __restrict__ VhT,  // [B,H,HD,L]
                                                   const float*  __restrict__ R,    // [B,H,L,8]
                                                   const float*  __restrict__ relv, // [5,HD]
                                                   bf16_t* __restrict__ ctx) {      // [B,L,HID]
    extern __shared__ char smem[];
    bf16_t* sP   = (bf16_t*)smem;                              // [16][L] scores -> probs
    float*  sOp  = (float*)(smem + 16 * L_ * sizeof(bf16_t));  // [8][16][64] wave partial O
    float*  sR   = sOp + 8 * 16 * 64;                          // [16][8] rel-k scores
    float*  sRV  = sR + 16 * 8;                                // [5][64] rel_v_table
    float*  sMax = sRV + 5 * 64;                               // [16]
    float*  sSum = sMax + 16;                                  // [16]
    float*  sRed = sSum + 16;                                  // [16][16] partial max/sum
    float*  sCp  = sRed + 256;                                 // [16][16][5] partial buckets
    float*  sC   = sCp + 16 * 16 * 5;                          // [16][5] bucket sums

    const int q0   = blockIdx.x * 16;
    const int h    = blockIdx.y;
    const int b    = blockIdx.z;
    const int tid  = threadIdx.x;
    const int wave = tid >> 5;
    const int lane = tid & 31;
    const int row  = lane & 15;
    const int half = lane >> 4;

    const bf16_t* Qbase = Qh  + (((size_t)b * H_ + h) * L_) * HD_;
    const bf16_t* Kbase = Kh  + (((size_t)b * H_ + h) * L_) * HD_;
    const bf16_t* Vbase = VhT + (((size_t)b * H_ + h) * HD_) * L_;
    const float*  Rbase = R   + (((size_t)b * H_ + h) * L_) * 8;

    if (tid < 128) sR[tid] = Rbase[(size_t)(q0 + (tid >> 3)) * 8 + (tid & 7)];
    for (int x = tid; x < 5 * 64; x += 256) sRV[x] = relv[x];
    __syncthreads();

    // ---- Pass 1: S = 0.5*(Q K^T + R[j(q,k)]) for this wave's 256 keys
    v16bf qa0, qa1;
    {
        const bf16_t* qp = Qbase + (size_t)(q0 + row) * HD_;
        qa0 = load_frag(qp, half);        // d = 0..31
        qa1 = load_frag(qp + 32, half);   // d = 32..63
    }
    const int kbeg = wave * (L_ / 8);
    for (int kk = kbeg; kk < kbeg + L_ / 8; kk += 16) {
        const bf16_t* kp = Kbase + (size_t)(kk + row) * HD_;  // B(c,n)=K[kk+n][c]
        v16bf kb0 = load_frag(kp, half);
        v16bf kb1 = load_frag(kp + 32, half);
        v8f s = {};
        s = wmma_bf16(qa0, kb0, s);
        s = wmma_bf16(qa1, kb1, s);
        const int kglob = kk + row;       // C layout: N = lane%16
#pragma unroll
        for (int i = 0; i < 8; ++i) {
            int qr = half * 8 + i;        // C layout: M = (lane/16)*8 + reg
            int dist = kglob - (q0 + qr);
            int j = dist < -2 ? 0 : (dist > 2 ? 4 : dist + 2);
            float val = 0.5f * (s[i] + sR[qr * 8 + j]);
            sP[(size_t)qr * L_ + kglob] = (bf16_t)val;
        }
    }
    __syncthreads();

    // ---- Pass 2: row max (16 threads per row, 128 cols each)
    {
        int r = tid >> 4, chunk = tid & 15;
        const bf16_t* pr = sP + (size_t)r * L_ + chunk * 128;
        float m = -3.0e38f;
        for (int x = 0; x < 128; ++x) m = fmaxf(m, (float)pr[x]);
        sRed[r * 16 + chunk] = m;
    }
    __syncthreads();
    if (tid < 16) {
        float m = -3.0e38f;
        for (int x = 0; x < 16; ++x) m = fmaxf(m, sRed[tid * 16 + x]);
        sMax[tid] = m;
    }
    __syncthreads();

    // ---- Pass 3: exp in place, row sums + 5 relative-position bucket sums
    {
        int r = tid >> 4, chunk = tid & 15;
        bf16_t* pr = sP + (size_t)r * L_ + chunk * 128;
        const int q = q0 + r, kstart = chunk * 128;
        const float m = sMax[r];
        float sum = 0.f, c0 = 0.f, c1 = 0.f, c2 = 0.f, c3 = 0.f, c4 = 0.f;
        for (int x = 0; x < 128; ++x) {
            float e = __expf((float)pr[x] - m);
            pr[x] = (bf16_t)e;
            sum += e;
            int dist = (kstart + x) - q;
            if      (dist <= -2) c0 += e;
            else if (dist == -1) c1 += e;
            else if (dist ==  0) c2 += e;
            else if (dist ==  1) c3 += e;
            else                 c4 += e;
        }
        sRed[r * 16 + chunk] = sum;
        float* cp = sCp + (r * 16 + chunk) * 5;
        cp[0] = c0; cp[1] = c1; cp[2] = c2; cp[3] = c3; cp[4] = c4;
    }
    __syncthreads();
    if (tid < 16) {
        float s = 0.f, c[5] = {0.f, 0.f, 0.f, 0.f, 0.f};
        for (int x = 0; x < 16; ++x) {
            s += sRed[tid * 16 + x];
            for (int j = 0; j < 5; ++j) c[j] += sCp[(tid * 16 + x) * 5 + j];
        }
        sSum[tid] = s;
        for (int j = 0; j < 5; ++j) sC[tid * 5 + j] = c[j];
    }
    __syncthreads();

    // ---- Pass 4: O_partial(wave) = P[:, kbeg:kbeg+256] * V, via WMMA from LDS
    v8f oacc[4] = {{}, {}, {}, {}};
    for (int kk = kbeg; kk < kbeg + L_ / 8; kk += 32) {
        const bf16_t* pp = sP + (size_t)row * L_ + kk;          // A: row = q row
        v16bf pa = load_frag(pp, half);
#pragma unroll
        for (int t = 0; t < 4; ++t) {
            // B(c,n) = V[kk+c][t*16+n] = VhT[t*16+n][kk+c] : contiguous per lane
            const bf16_t* vp = Vbase + (size_t)(t * 16 + row) * L_ + kk;
            v16bf vb = load_frag(vp, half);
            oacc[t] = wmma_bf16(pa, vb, oacc[t]);
        }
    }
#pragma unroll
    for (int t = 0; t < 4; ++t)
#pragma unroll
        for (int i = 0; i < 8; ++i)
            sOp[((wave * 16) + half * 8 + i) * 64 + t * 16 + row] = oacc[t][i];
    __syncthreads();

    // ---- Pass 5: deterministic reduce + rel-v contribution + normalize
    for (int x = tid; x < 16 * 64; x += 256) {
        int r = x >> 6, d = x & 63;
        float val = 0.f;
        for (int w = 0; w < 8; ++w) val += sOp[(w * 16 + r) * 64 + d];
        for (int j = 0; j < 5; ++j) val += sC[r * 5 + j] * sRV[j * 64 + d];
        val /= sSum[r];
        ctx[((size_t)b * L_ + q0 + r) * HID_ + (size_t)h * HD_ + d] = (bf16_t)val;
    }
}

// ---------------------------------------------------------------- host launcher
extern "C" void kernel_launch(void* const* d_in, const int* in_sizes, int n_in,
                              void* d_out, int out_size, void* d_ws, size_t ws_size,
                              hipStream_t stream) {
    const float* query = (const float*)d_in[0];
    const float* key   = (const float*)d_in[1];
    const float* value = (const float*)d_in[2];
    const float* Wq = (const float*)d_in[3];
    const float* bq = (const float*)d_in[4];
    const float* Wk = (const float*)d_in[5];
    const float* bk = (const float*)d_in[6];
    const float* Wv = (const float*)d_in[7];
    const float* bv = (const float*)d_in[8];
    const float* Wo = (const float*)d_in[9];
    const float* bo = (const float*)d_in[10];
    const float* relk = (const float*)d_in[11];
    const float* relv = (const float*)d_in[12];

    const size_t nX = (size_t)B_ * L_ * HID_;   // 4,194,304 activation elements
    const size_t nW = (size_t)HID_ * HID_;      // 1,048,576 weight elements

    char* ws = (char*)d_ws;
    size_t off = 0;
    auto carve = [&](size_t bytes) {
        char* p = ws + off;
        off += (bytes + 255) & ~(size_t)255;
        return p;
    };
    bf16_t* xq  = (bf16_t*)carve(nX * 2);
    bf16_t* xk  = (bf16_t*)carve(nX * 2);
    bf16_t* xv  = (bf16_t*)carve(nX * 2);
    bf16_t* wq  = (bf16_t*)carve(nW * 2);
    bf16_t* wk  = (bf16_t*)carve(nW * 2);
    bf16_t* wv  = (bf16_t*)carve(nW * 2);
    bf16_t* wo  = (bf16_t*)carve(nW * 2);
    bf16_t* qh  = (bf16_t*)carve(nX * 2);       // [B,H,L,HD]
    bf16_t* kh  = (bf16_t*)carve(nX * 2);       // [B,H,L,HD]
    bf16_t* vhT = (bf16_t*)carve(nX * 2);       // [B,H,HD,L]
    float*  Rj  = (float*) carve((size_t)B_ * H_ * L_ * 8 * 4);  // [B,H,L,8]
    bf16_t* ctx = (bf16_t*)carve(nX * 2);       // [B,L,HID]
    (void)ws_size; (void)in_sizes; (void)n_in; (void)out_size;

    // 1) f32 -> bf16
    dim3 cvb(256);
    f32_to_bf16_kernel<<<dim3((nX + 255) / 256), cvb, 0, stream>>>(query, xq, nX);
    f32_to_bf16_kernel<<<dim3((nX + 255) / 256), cvb, 0, stream>>>(key,   xk, nX);
    f32_to_bf16_kernel<<<dim3((nX + 255) / 256), cvb, 0, stream>>>(value, xv, nX);
    f32_to_bf16_kernel<<<dim3((nW + 255) / 256), cvb, 0, stream>>>(Wq, wq, nW);
    f32_to_bf16_kernel<<<dim3((nW + 255) / 256), cvb, 0, stream>>>(Wk, wk, nW);
    f32_to_bf16_kernel<<<dim3((nW + 255) / 256), cvb, 0, stream>>>(Wv, wv, nW);
    f32_to_bf16_kernel<<<dim3((nW + 255) / 256), cvb, 0, stream>>>(Wo, wo, nW);

    // 2) Q/K/V projections: async-to-LDS double-buffered WMMA GEMM
    dim3 gg(HID_ / NT, (B_ * L_) / MT);   // 8 x 32 macro-tiles
    tiled_gemm_kernel<<<gg, dim3(256), GEMM_SMEM, stream>>>(xq, wq, bq, qh,  HID_, 0);
    tiled_gemm_kernel<<<gg, dim3(256), GEMM_SMEM, stream>>>(xk, wk, bk, kh,  HID_, 0);
    tiled_gemm_kernel<<<gg, dim3(256), GEMM_SMEM, stream>>>(xv, wv, bv, vhT, HID_, 1);

    // 3) 5-entry relative-key score table
    {
        size_t total = (size_t)B_ * H_ * L_ * 5;
        relk_kernel<<<dim3((total + 255) / 256), dim3(256), 0, stream>>>(qh, relk, Rj);
    }

    // 4) fused attention (scores + rel + softmax + buckets + PV + rel-v) -> ctx bf16
    attn_kernel<<<dim3(L_ / 16, H_, B_), dim3(256), SMEM_BYTES, stream>>>(
        qh, kh, vhT, Rj, relv, ctx);

    // 5) output projection -> f32 d_out
    tiled_gemm_kernel<<<gg, dim3(256), GEMM_SMEM, stream>>>(ctx, wo, bo, d_out, HID_, 2);
}
